// LowRankSS2D_88983132438829
// MI455X (gfx1250) — matrix operand order repaired and checked by reading, not compiled, not run on hardware
//
#include <hip/hip_runtime.h>
#include <hip/hip_bf16.h>

// ---------------------------------------------------------------------------
// LowRankSS2D for gfx1250 (MI455X): fused input-proj GEMM + exact blocked
// parallel linear scan (64 segments x 64 steps, A^64 = U G^63 V^T combine)
// + fused output-proj / residual / LayerNorm. All matmuls via
// v_wmma_f32_16x16x32_f16 (fp32 accumulate).
// ---------------------------------------------------------------------------

typedef __attribute__((ext_vector_type(16))) _Float16 v16h;
typedef __attribute__((ext_vector_type(8)))  _Float16 v8h;
typedef __attribute__((ext_vector_type(8)))  float    v8f;

#define WMMA(a, b, c) \
  __builtin_amdgcn_wmma_f32_16x16x32_f16(false, (a), false, (b), (short)0, (c), false, false)

// A operand (16x32 f16, dense-A layout): lane m = lane&15, kh = lane>>4.
// elems 0..7  <- row[k0 + kh*8 + j]        (16B contiguous -> ds/global b128)
// elems 8..15 <- row[k0 + 16 + kh*8 + j]
__device__ __forceinline__ v16h packA(const _Float16* row, int k0, int kh) {
  v16h a;
  ((v8h*)&a)[0] = *(const v8h*)(row + k0 + kh * 8);
  ((v8h*)&a)[1] = *(const v8h*)(row + k0 + 16 + kh * 8);
  return a;
}

// B operand (32x16 f16): lane n = lane&15; same k order as packA:
// elems 0..7 <- rowN[k0 + kh*8 + j], elems 8..15 <- rowN[k0 + 16 + kh*8 + j].
__device__ __forceinline__ v16h packB(const _Float16* rowN, int k0, int kh) {
  v16h b;
  ((v8h*)&b)[0] = *(const v8h*)(rowN + k0 + kh * 8);
  ((v8h*)&b)[1] = *(const v8h*)(rowN + k0 + 16 + kh * 8);
  return b;
}

// ---------------------------------------------------------------------------
// prep: Weff = Wb@Wi (f16), beff = Wb@bi + bb (f32), f16 copies of Wo, Vt, U
// ---------------------------------------------------------------------------
__global__ void prep_kernel(const float* Wi, const float* bi, const float* Wb,
                            const float* bb, const float* Wo, const float* U,
                            const float* V, _Float16* WeffH, float* beff,
                            _Float16* WoH, _Float16* VtH, _Float16* UH) {
  int id = blockIdx.x * blockDim.x + threadIdx.x;
  if (id < 32768) {                       // Weff[n][c] = sum_s Wb[n][s]*Wi[s][c]
    int n = id >> 8, k = id & 255;
    float s = 0.f;
    for (int j = 0; j < 128; ++j) s += Wb[n * 128 + j] * Wi[j * 256 + k];
    WeffH[n * 256 + k] = (_Float16)s;
  } else if (id < 65536) {                // Wo f16 copy [256][128]
    int i = id - 32768;
    WoH[i] = (_Float16)Wo[i];
  } else if (id < 73728) {                // Vt[r][s] = V[s][r]  -> [64][128]
    int i = id - 65536;
    int n = i >> 7, k = i & 127;
    VtH[n * 128 + k] = (_Float16)V[k * 64 + n];
  } else if (id < 81920) {                // U f16 copy [128][64]
    int i = id - 73728;
    UH[i] = (_Float16)U[i];
  } else if (id < 82048) {                // beff[n]
    int n = id - 81920;
    float s = 0.f;
    for (int j = 0; j < 128; ++j) s += Wb[n * 128 + j] * bi[j];
    beff[n] = s + bb[n];
  }
}

// ---------------------------------------------------------------------------
// G63: G = V^T U (64x64), then G^63 by repeated squaring; stored f16 row-major
// ---------------------------------------------------------------------------
__global__ void g63_kernel(const float* U, const float* V, _Float16* GH) {
  __shared__ float GA[64][65];
  __shared__ float GB[64][65];
  __shared__ float GC[64][65];
  const int tid = threadIdx.x;
  for (int e = 0; e < 16; ++e) {
    int idx = tid * 16 + e;
    int i = idx >> 6, j = idx & 63;
    float s = 0.f;
    for (int ss = 0; ss < 128; ++ss) s += V[ss * 64 + i] * U[ss * 64 + j];
    GA[i][j] = s;
    GB[i][j] = s;
  }
  __syncthreads();
  for (int it = 0; it < 5; ++it) {
    for (int e = 0; e < 16; ++e) {        // GC = GA*GA
      int idx = tid * 16 + e; int i = idx >> 6, j = idx & 63;
      float s = 0.f;
      for (int k = 0; k < 64; ++k) s += GA[i][k] * GA[k][j];
      GC[i][j] = s;
    }
    __syncthreads();
    for (int e = 0; e < 16; ++e) { int idx = tid * 16 + e; GA[idx >> 6][idx & 63] = GC[idx >> 6][idx & 63]; }
    __syncthreads();
    for (int e = 0; e < 16; ++e) {        // GC = GB*GA
      int idx = tid * 16 + e; int i = idx >> 6, j = idx & 63;
      float s = 0.f;
      for (int k = 0; k < 64; ++k) s += GB[i][k] * GA[k][j];
      GC[i][j] = s;
    }
    __syncthreads();
    for (int e = 0; e < 16; ++e) { int idx = tid * 16 + e; GB[idx >> 6][idx & 63] = GC[idx >> 6][idx & 63]; }
    __syncthreads();
  }
  // GB = G^63  (G * G^2 * G^4 * G^8 * G^16 * G^32)
  for (int e = 0; e < 16; ++e) { int idx = tid * 16 + e; GH[idx] = (_Float16)GB[idx >> 6][idx & 63]; }
}

// ---------------------------------------------------------------------------
// bx GEMM: bx[l][b][s] = seq[b][l][:] @ Weff^T + beff  (M=B*L, N=128, K=256)
// one block = 16 consecutive l at fixed b; 4 waves x 2 N-tiles
// ---------------------------------------------------------------------------
__global__ void bx_gemm_kernel(const float* x, const _Float16* WeffH,
                               const float* beff, float* bx) {
  __shared__ __align__(16) _Float16 Abf[16][264];  // m-major tile, padded
  const int bid = blockIdx.x;                      // 0..4095
  const int b = bid >> 8;
  const int l0 = (bid & 255) << 4;
  const int tid = threadIdx.x;                     // 0..127
  for (int c = tid; c < 256; c += 128) {           // seq[b][l0+m][c] = x[b][c][l0+m]
    const float* src = x + (((size_t)b * 256 + c) << 12) + l0;
#pragma unroll
    for (int q = 0; q < 4; ++q) {
      float4 v = *(const float4*)(src + q * 4);
      Abf[q * 4 + 0][c] = (_Float16)v.x;
      Abf[q * 4 + 1][c] = (_Float16)v.y;
      Abf[q * 4 + 2][c] = (_Float16)v.z;
      Abf[q * 4 + 3][c] = (_Float16)v.w;
    }
  }
  __syncthreads();
  const int wv = tid >> 5, lane = tid & 31, ln = lane & 15, kh = lane >> 4;
  const _Float16* arow = &Abf[ln][0];
#pragma unroll
  for (int tile = 0; tile < 2; ++tile) {
    const int n0 = tile * 64 + wv * 16;
    const _Float16* wrow = WeffH + (size_t)(n0 + ln) * 256;
    v8f c = {};
#pragma unroll
    for (int ks = 0; ks < 8; ++ks) {
      v16h a = packA(arow, ks * 32, kh);
      v16h bm = packB(wrow, ks * 32, kh);
      c = WMMA(a, bm, c);
    }
    const float bias = beff[n0 + ln];
#pragma unroll
    for (int r = 0; r < 8; ++r) {
      const int l = l0 + kh * 8 + r;
      bx[(((size_t)l * 16 + b) << 7) + n0 + ln] = c[r] + bias;
    }
  }
}

// ---------------------------------------------------------------------------
// scan: h[16,128] over a 64-step segment. GEMM1 (waves 0-3): tmp = h@V;
// GEMM2 (8 waves): h = tmp@U^T + bx_t. pass1: zero init -> rseg.
// pass2: init from carry, write outputs to hsA/hsB.
// ---------------------------------------------------------------------------
__global__ void scan_kernel(int pass, const float* bx, const _Float16* VtH,
                            const _Float16* UH, float* rseg, const float* carry,
                            float* hsA, float* hsB) {
  __shared__ __align__(16) _Float16 Hb[16][136];
  __shared__ __align__(16) _Float16 Tb[16][72];
  const int dir = blockIdx.x & 1;
  const int seg = blockIdx.x >> 1;
  const int tid = threadIdx.x;
  const int wv = tid >> 5, lane = tid & 31, ln = lane & 15, kh = lane >> 4;
  v16h Vop[4], Uop[2];
  if (wv < 4) {
    const _Float16* vrow = VtH + (size_t)(wv * 16 + ln) * 128;
#pragma unroll
    for (int ks = 0; ks < 4; ++ks) Vop[ks] = packB(vrow, ks * 32, kh);
  }
  const int n0u = wv * 16;
  {
    const _Float16* urow = UH + (size_t)(n0u + ln) * 64;
#pragma unroll
    for (int ks = 0; ks < 2; ++ks) Uop[ks] = packB(urow, ks * 32, kh);
  }
  if (pass == 1) {
    for (int i = tid; i < 16 * 136; i += 256) (&Hb[0][0])[i] = (_Float16)0.f;
  } else {
    const float* cp = carry + (((size_t)(dir * 64 + seg) * 16) << 7);
    for (int i = tid; i < 2048; i += 256) Hb[i >> 7][i & 127] = (_Float16)cp[i];
  }
  __syncthreads();
  const _Float16* hrow = &Hb[ln][0];
  const _Float16* trow = &Tb[ln][0];
  v8f h = {};
  for (int tt = 0; tt < 64; ++tt) {
    const int tg = seg * 64 + tt;
    const int li = dir ? (4095 - tg) : tg;
    if (wv < 4) {
      v8f c1 = {};
#pragma unroll
      for (int ks = 0; ks < 4; ++ks) {
        v16h a = packA(hrow, ks * 32, kh);
        c1 = WMMA(a, Vop[ks], c1);
      }
#pragma unroll
      for (int r = 0; r < 8; ++r) Tb[kh * 8 + r][wv * 16 + ln] = (_Float16)c1[r];
    } else if (tt + 1 < 64) {           // waves 4-7: prefetch next bx row (8KB)
      const int liN = dir ? (4094 - tg) : (tg + 1);
      const float* pf = bx + (((size_t)liN * 16) << 7) + ((wv - 4) * 32 + lane) * 16;
      __builtin_prefetch(pf, 0, 0);
    }
    __syncthreads();
    const float* bxp = bx + (((size_t)li * 16) << 7);
#pragma unroll
    for (int r = 0; r < 8; ++r) h[r] = bxp[((kh * 8 + r) << 7) + n0u + ln];
#pragma unroll
    for (int ks = 0; ks < 2; ++ks) {
      v16h a = packA(trow, ks * 32, kh);
      h = WMMA(a, Uop[ks], h);
    }
    if (pass == 2) {
      float* hp = (dir ? hsB : hsA) + (((size_t)li * 16) << 7);
#pragma unroll
      for (int r = 0; r < 8; ++r) hp[((kh * 8 + r) << 7) + n0u + ln] = h[r];
    }
#pragma unroll
    for (int r = 0; r < 8; ++r) Hb[kh * 8 + r][n0u + ln] = (_Float16)h[r];
    __syncthreads();
  }
  if (pass == 1) {
    float* rp = rseg + (((size_t)(dir * 64 + seg) * 16) << 7);
#pragma unroll
    for (int r = 0; r < 8; ++r) rp[((kh * 8 + r) << 7) + n0u + ln] = h[r];
  }
}

// ---------------------------------------------------------------------------
// combine: c_0 = 0; c_{j+1} = c_j * (V G63^T U^T) + r_j ; stores each c_j.
// ---------------------------------------------------------------------------
__global__ void combine_kernel(const _Float16* VtH, const _Float16* UH,
                               const _Float16* GH, const float* rseg,
                               float* carry) {
  __shared__ __align__(16) _Float16 Hb[16][136];
  __shared__ __align__(16) _Float16 Tb[16][72];
  __shared__ __align__(16) _Float16 T2b[16][72];
  const int dir = blockIdx.x;
  const int tid = threadIdx.x;
  const int wv = tid >> 5, lane = tid & 31, ln = lane & 15, kh = lane >> 4;
  v16h Vop[4], Gop[2], Uop[2];
  if (wv < 4) {
    const _Float16* vrow = VtH + (size_t)(wv * 16 + ln) * 128;
#pragma unroll
    for (int ks = 0; ks < 4; ++ks) Vop[ks] = packB(vrow, ks * 32, kh);
    const _Float16* grow = GH + (size_t)(wv * 16 + ln) * 64;
#pragma unroll
    for (int ks = 0; ks < 2; ++ks) Gop[ks] = packB(grow, ks * 32, kh);
  }
  {
    const _Float16* urow = UH + (size_t)(wv * 16 + ln) * 64;
#pragma unroll
    for (int ks = 0; ks < 2; ++ks) Uop[ks] = packB(urow, ks * 32, kh);
  }
  for (int i = tid; i < 16 * 136; i += 256) (&Hb[0][0])[i] = (_Float16)0.f;
  __syncthreads();
  const _Float16* hrow = &Hb[ln][0];
  const _Float16* trow = &Tb[ln][0];
  const _Float16* t2row = &T2b[ln][0];
  for (int seg = 0; seg < 64; ++seg) {
    float* cp = carry + (((size_t)(dir * 64 + seg) * 16) << 7);
    for (int i = tid; i < 2048; i += 256) cp[i] = (float)Hb[i >> 7][i & 127];
    if (wv < 4) {
      v8f c1 = {};
#pragma unroll
      for (int ks = 0; ks < 4; ++ks) {
        v16h a = packA(hrow, ks * 32, kh);
        c1 = WMMA(a, Vop[ks], c1);
      }
#pragma unroll
      for (int r = 0; r < 8; ++r) Tb[kh * 8 + r][wv * 16 + ln] = (_Float16)c1[r];
    }
    __syncthreads();
    if (wv < 4) {
      v8f cm = {};
#pragma unroll
      for (int ks = 0; ks < 2; ++ks) {
        v16h a = packA(trow, ks * 32, kh);
        cm = WMMA(a, Gop[ks], cm);
      }
#pragma unroll
      for (int r = 0; r < 8; ++r) T2b[kh * 8 + r][wv * 16 + ln] = (_Float16)cm[r];
    }
    __syncthreads();
    const float* rp = rseg + (((size_t)(dir * 64 + seg) * 16) << 7);
    v8f h;
#pragma unroll
    for (int r = 0; r < 8; ++r) h[r] = rp[((kh * 8 + r) << 7) + wv * 16 + ln];
#pragma unroll
    for (int ks = 0; ks < 2; ++ks) {
      v16h a = packA(t2row, ks * 32, kh);
      h = WMMA(a, Uop[ks], h);
    }
#pragma unroll
    for (int r = 0; r < 8; ++r) Hb[kh * 8 + r][wv * 16 + ln] = (_Float16)h[r];
    __syncthreads();
  }
}

// ---------------------------------------------------------------------------
// output projection + residual + LayerNorm. One block per l (16 b-rows).
// y_lr = hsA@Wo^T+bo ; y = LN(0.5*(hsA+hsB)@Wo^T+bo + seq)*gamma+beta
// ---------------------------------------------------------------------------
__global__ void outln_kernel(const float* hsA, const float* hsB,
                             const _Float16* WoH, const float* bo,
                             const float* x, const float* gamma,
                             const float* beta, float* out_y, float* out_ylr) {
  __shared__ __align__(16) _Float16 A1b[16][136];
  __shared__ __align__(16) _Float16 A2b[16][136];
  __shared__ float Ys[16][260];
  const int l = blockIdx.x;
  const int tid = threadIdx.x;
  {
    const int idx = tid * 8;
    const int m = idx >> 7, k = idx & 127;
    const float* pa = hsA + (((size_t)l * 16 + m) << 7) + k;
    const float* pb = hsB + (((size_t)l * 16 + m) << 7) + k;
    float4 a0 = ((const float4*)pa)[0], a1 = ((const float4*)pa)[1];
    float4 b0 = ((const float4*)pb)[0], b1 = ((const float4*)pb)[1];
    float av[8] = {a0.x, a0.y, a0.z, a0.w, a1.x, a1.y, a1.z, a1.w};
    float bv[8] = {b0.x, b0.y, b0.z, b0.w, b1.x, b1.y, b1.z, b1.w};
#pragma unroll
    for (int e = 0; e < 8; ++e) {
      A1b[m][k + e] = (_Float16)av[e];
      A2b[m][k + e] = (_Float16)(0.5f * (av[e] + bv[e]));
    }
  }
  __syncthreads();
  const int wv = tid >> 5, lane = tid & 31, ln = lane & 15, kh = lane >> 4;
  const _Float16* a1row = &A1b[ln][0];
  const _Float16* a2row = &A2b[ln][0];
#pragma unroll
  for (int tile = 0; tile < 2; ++tile) {
    const int n0 = wv * 32 + tile * 16;
    const _Float16* wrow = WoH + (size_t)(n0 + ln) * 128;
    v8f cL = {}, cS = {};
#pragma unroll
    for (int ks = 0; ks < 4; ++ks) {
      v16h bm = packB(wrow, ks * 32, kh);
      cL = WMMA(packA(a1row, ks * 32, kh), bm, cL);
      cS = WMMA(packA(a2row, ks * 32, kh), bm, cS);
    }
    const int cc = n0 + ln;
    const float bias = bo[cc];
#pragma unroll
    for (int r = 0; r < 8; ++r) {
      const int br = kh * 8 + r;
      out_ylr[(((size_t)br * 4096 + l) << 8) + cc] = cL[r] + bias;
      const float res = x[(((size_t)br * 256 + cc) << 12) + l];
      Ys[br][cc] = cS[r] + bias + res;
    }
  }
  __syncthreads();
  const int row = tid >> 4, j = tid & 15;
  float s = 0.f, s2 = 0.f;
#pragma unroll
  for (int i = 0; i < 16; ++i) {
    const float vv = Ys[row][j + i * 16];
    s += vv;
    s2 += vv * vv;
  }
#pragma unroll
  for (int off = 1; off < 16; off <<= 1) {
    s += __shfl_xor(s, off, 32);
    s2 += __shfl_xor(s2, off, 32);
  }
  const float mean = s * (1.f / 256.f);
  const float var = s2 * (1.f / 256.f) - mean * mean;
  const float inv = rsqrtf(var + 1e-5f);
#pragma unroll
  for (int i = 0; i < 16; ++i) {
    const int c = j + i * 16;
    const float vv = (Ys[row][c] - mean) * inv * gamma[c] + beta[c];
    out_y[(((size_t)row * 256 + c) << 12) + l] = vv;
  }
}

// ---------------------------------------------------------------------------
extern "C" void kernel_launch(void* const* d_in, const int* in_sizes, int n_in,
                              void* d_out, int out_size, void* d_ws, size_t ws_size,
                              hipStream_t stream) {
  (void)in_sizes; (void)n_in; (void)out_size; (void)ws_size;
  const float* x     = (const float*)d_in[0];
  const float* Wi    = (const float*)d_in[1];
  const float* bi    = (const float*)d_in[2];
  const float* U     = (const float*)d_in[3];
  const float* V     = (const float*)d_in[4];
  const float* Wb    = (const float*)d_in[5];
  const float* bb    = (const float*)d_in[6];
  const float* Wo    = (const float*)d_in[7];
  const float* bo    = (const float*)d_in[8];
  const float* gamma = (const float*)d_in[9];
  const float* beta  = (const float*)d_in[10];

  // workspace layout (~103 MB)
  float* bx    = (float*)d_ws;            // [4096][16][128] f32
  float* hsA   = bx + 8388608;            // [4096][16][128] f32
  float* hsB   = hsA + 8388608;           // [4096][16][128] f32
  float* rseg  = hsB + 8388608;           // [2][64][16][128] f32
  float* carry = rseg + 262144;           // [2][64][16][128] f32
  float* beff  = carry + 262144;          // [128] f32
  _Float16* WeffH = (_Float16*)(beff + 128);  // [128][256]
  _Float16* WoH   = WeffH + 32768;            // [256][128]
  _Float16* VtH   = WoH + 32768;              // [64][128]
  _Float16* UH    = VtH + 8192;               // [128][64]
  _Float16* GH    = UH + 8192;                // [64][64]

  float* out_y   = (float*)d_out;         // [16][256][64][64]
  float* out_ylr = out_y + 16777216;      // [16][4096][256]

  prep_kernel<<<321, 256, 0, stream>>>(Wi, bi, Wb, bb, Wo, U, V, WeffH, beff, WoH, VtH, UH);
  g63_kernel<<<1, 256, 0, stream>>>(U, V, GH);
  bx_gemm_kernel<<<4096, 128, 0, stream>>>(x, WeffH, beff, bx);
  scan_kernel<<<128, 256, 0, stream>>>(1, bx, VtH, UH, rseg, carry, hsA, hsB);
  combine_kernel<<<2, 256, 0, stream>>>(VtH, UH, GH, rseg, carry);
  scan_kernel<<<128, 256, 0, stream>>>(2, bx, VtH, UH, rseg, carry, hsA, hsB);
  outln_kernel<<<4096, 256, 0, stream>>>(hsA, hsB, WoH, bo, x, gamma, beta, out_y, out_ylr);
}
